// WubuBlock_22024592294279
// MI455X (gfx1250) — matrix operand order repaired and checked by reading, not compiled, not run on hardware
//
#include <hip/hip_runtime.h>
#include <math.h>

#define DIMX   512
#define HEADS  8
#define HDIM   64
#define KNN    32
#define NROWS  4096
#define EPSF   1e-7f

typedef __bf16 bf16_t;
typedef bf16_t bf16x8  __attribute__((ext_vector_type(8)));
typedef bf16_t bf16x16 __attribute__((ext_vector_type(16)));
typedef float  floatx8 __attribute__((ext_vector_type(8)));

__device__ __forceinline__ float gelu_tanh(float v) {
  float v3 = v * v * v;
  return 0.5f * v * (1.0f + tanhf(0.7978845608028654f * (v + 0.044715f * v3)));
}

// ---------------------------------------------------------------------------
// LayerNorm: one block (256 threads) per row of 512.
// ---------------------------------------------------------------------------
__global__ __launch_bounds__(256) void ln_rows(const float* __restrict__ x,
                                               const float* __restrict__ s,
                                               const float* __restrict__ b,
                                               float* __restrict__ out) {
  int row = blockIdx.x, tid = threadIdx.x;
  const float* xr = x + (size_t)row * DIMX;
  float e0 = xr[tid], e1 = xr[tid + 256];
  float sum = e0 + e1, sq = e0 * e0 + e1 * e1;
#pragma unroll
  for (int off = 16; off > 0; off >>= 1) {
    sum += __shfl_xor(sum, off, 32);
    sq  += __shfl_xor(sq,  off, 32);
  }
  __shared__ float ssum[8], ssq[8];
  if ((tid & 31) == 0) { ssum[tid >> 5] = sum; ssq[tid >> 5] = sq; }
  __syncthreads();
  float ts = 0.f, tq = 0.f;
#pragma unroll
  for (int i = 0; i < 8; i++) { ts += ssum[i]; tq += ssq[i]; }
  float mean = ts * (1.0f / DIMX);
  float var  = tq * (1.0f / DIMX) - mean * mean;
  float inv  = rsqrtf(var + 1e-6f);
  float* orow = out + (size_t)row * DIMX;
  orow[tid]       = (e0 - mean) * inv * s[tid]       + b[tid];
  orow[tid + 256] = (e1 - mean) * inv * s[tid + 256] + b[tid + 256];
}

// ---------------------------------------------------------------------------
// bf16 WMMA GEMM: C[M,N] = epilogue(A[M,K] @ B[K,N] + bias [+ resid]).
// Block 256 = 8 waves (4 x 2); workgroup tile 128x64; wave tile 32x32
// (2x2 of 16x16x32 WMMA). Double-buffered LDS with register staging:
// global loads for tile t+1 are issued during WMMA compute of tile t;
// one barrier per K-step. mode: 0 = +bias, 1 = +bias+resid, 2 = gelu(+bias)
// ---------------------------------------------------------------------------
__global__ __launch_bounds__(256) void gemm_wmma_bf16(
    const float* __restrict__ A, const float* __restrict__ B,
    const float* __restrict__ bias, const float* __restrict__ resid,
    float* __restrict__ C, int Kdim, int Ndim, int mode) {
  __shared__ bf16_t lA[2][128 * 32];   // [buf][row 0..127][k 0..31]
  __shared__ bf16_t lB[2][64 * 32];    // [buf][n 0..63][k 0..31] (transposed)

  int tid  = threadIdx.x;
  int lane = tid & 31;
  int wave = tid >> 5;
  int wm = wave & 3, wn = wave >> 2;
  int m0 = blockIdx.y * 128, n0 = blockIdx.x * 64;
  int hs  = (lane >> 4) & 1;   // half-wave select
  int l16 = lane & 15;

  floatx8 acc[2][2];
#pragma unroll
  for (int i = 0; i < 2; i++)
#pragma unroll
    for (int j = 0; j < 2; j++)
#pragma unroll
      for (int r = 0; r < 8; r++) acc[i][j][r] = 0.0f;

  // staging thread mapping
  int ar  = tid >> 3;          // A row 0..31 (+ rr*32)
  int ac  = (tid & 7) * 4;     // A col 0..28 step 4
  int bn4 = (tid & 15) * 4;    // B col group (4 cols)
  int bkr = tid >> 4;          // B k row 0..15 (+ u*16)

  float4 aReg[4];
  float4 bReg[2];

  auto load_tile = [&](int k0) {
#pragma unroll
    for (int rr = 0; rr < 4; rr++)
      aReg[rr] = *(const float4*)&A[(size_t)(m0 + ar + rr * 32) * Kdim + k0 + ac];
    bReg[0] = *(const float4*)&B[(size_t)(k0 + bkr) * Ndim + n0 + bn4];
    bReg[1] = *(const float4*)&B[(size_t)(k0 + bkr + 16) * Ndim + n0 + bn4];
  };
  auto store_tile = [&](int p) {
#pragma unroll
    for (int rr = 0; rr < 4; rr++) {
      bf16_t* dst = &lA[p][(ar + rr * 32) * 32 + ac];
      dst[0] = (bf16_t)aReg[rr].x; dst[1] = (bf16_t)aReg[rr].y;
      dst[2] = (bf16_t)aReg[rr].z; dst[3] = (bf16_t)aReg[rr].w;
    }
#pragma unroll
    for (int u = 0; u < 2; u++) {
      int kk = bkr + u * 16;
      float4 v = bReg[u];
      lB[p][(bn4 + 0) * 32 + kk] = (bf16_t)v.x;
      lB[p][(bn4 + 1) * 32 + kk] = (bf16_t)v.y;
      lB[p][(bn4 + 2) * 32 + kk] = (bf16_t)v.z;
      lB[p][(bn4 + 3) * 32 + kk] = (bf16_t)v.w;
    }
  };

  int nT = Kdim >> 5;
  load_tile(0);
  int p = 0;
  for (int t = 0; t < nT; t++) {
    store_tile(p);
    __syncthreads();                 // tile p visible; also fences buffer reuse
    if (t + 1 < nT) {
      load_tile((t + 1) * 32);       // in flight during WMMA compute below
      if (t + 2 < nT)
        __builtin_prefetch(&A[(size_t)(m0 + ar) * Kdim + (t + 2) * 32 + ac], 0, 1);
    }
    // Fragment loads per ISA 16-bit layouts:
    // A lane l: row = l%16; halfs[0..7] = K (hs?8:0)+0..7, halfs[8..15] = +16
    // B lane l: col = l%16; halfs[0..7] = K (hs?8:0)+0..7, halfs[8..15] = +16
    bf16x16 afr[2], bfr[2];
#pragma unroll
    for (int s = 0; s < 2; s++) {
      const bf16_t* ap = &lA[p][(wm * 32 + s * 16 + l16) * 32 + hs * 8];
      bf16x8 lo = *(const bf16x8*)ap;
      bf16x8 hi = *(const bf16x8*)(ap + 16);
#pragma unroll
      for (int i = 0; i < 8; i++) { afr[s][i] = lo[i]; afr[s][8 + i] = hi[i]; }
      const bf16_t* bp = &lB[p][(wn * 32 + s * 16 + l16) * 32 + hs * 8];
      bf16x8 blo = *(const bf16x8*)bp;
      bf16x8 bhi = *(const bf16x8*)(bp + 16);
#pragma unroll
      for (int i = 0; i < 8; i++) { bfr[s][i] = blo[i]; bfr[s][8 + i] = bhi[i]; }
    }
#pragma unroll
    for (int i = 0; i < 2; i++)
#pragma unroll
      for (int j = 0; j < 2; j++)
        acc[i][j] = __builtin_amdgcn_wmma_f32_16x16x32_bf16(
            false, afr[i], false, bfr[j], (short)0, acc[i][j], false, false);
    p ^= 1;
  }

  // Epilogue. C layout: VGPR r -> M = base + (hs?8:0) + r, N = base + l%16.
#pragma unroll
  for (int j = 0; j < 2; j++) {
    int n = n0 + wn * 32 + j * 16 + l16;
    float bval = bias[n];
#pragma unroll
    for (int i = 0; i < 2; i++) {
      int mb = m0 + wm * 32 + i * 16 + hs * 8;
#pragma unroll
      for (int r = 0; r < 8; r++) {
        int m = mb + r;
        float v = acc[i][j][r] + bval;
        if (mode == 2) v = gelu_tanh(v);
        if (mode == 1) v += resid[(size_t)m * Ndim + n];
        C[(size_t)m * Ndim + n] = v;
      }
    }
  }
}

// ---------------------------------------------------------------------------
// Top-K nearest in hyperbolic metric. dist() is monotone in ||mobius_diff||,
// so the ranking key is ||num||^2/den^2 -- no arctanh needed.
// Block 128 = 4 waves, one row per wave; row keys live in LDS (64 KB/block).
// ---------------------------------------------------------------------------
__global__ __launch_bounds__(128) void topk_knn(const float* __restrict__ pos,
                                                const float* __restrict__ cptr,
                                                int* __restrict__ knn) {
  __shared__ float sd[4][NROWS];
  int tid = threadIdx.x, lane = tid & 31, wid = tid >> 5;
  int row = blockIdx.x * 4 + wid;
  float c  = *cptr;
  float ix = pos[row * 3 + 0], iy = pos[row * 3 + 1], iz = pos[row * 3 + 2];
  float i2 = ix * ix + iy * iy + iz * iz;
  float* sdw = sd[wid];

  for (int t = 0; t < NROWS / 32; t++) {
    int j = lane + t * 32;
    float jx = pos[j * 3 + 0], jy = pos[j * 3 + 1], jz = pos[j * 3 + 2];
    float j2 = jx * jx + jy * jy + jz * jz;
    float xy = -(jx * ix + jy * iy + jz * iz);   // x=p_j, y=-p_i
    float a  = 1.f + 2.f * c * xy + c * i2;
    float bb = 1.f - c * j2;
    float nx = a * jx - bb * ix;
    float ny = a * jy - bb * iy;
    float nz = a * jz - bb * iz;
    float den = fmaxf(1.f + 2.f * c * xy + c * c * j2 * i2, EPSF);
    sdw[j] = (nx * nx + ny * ny + nz * nz) / (den * den);
  }
  __syncthreads();

  for (int kk = 0; kk < KNN; kk++) {
    float best = 3.4e38f;
    int   bj   = 1 << 30;
    for (int t = 0; t < NROWS / 32; t++) {
      int j = lane + t * 32;                 // ascending -> ties keep low j
      float v = sdw[j];
      if (v < best) { best = v; bj = j; }
    }
#pragma unroll
    for (int off = 16; off > 0; off >>= 1) { // argmin, tie -> smaller index
      float ov = __shfl_xor(best, off, 32);
      int   oj = __shfl_xor(bj,   off, 32);
      if (ov < best || (ov == best && oj < bj)) { best = ov; bj = oj; }
    }
    if (lane == 0) { knn[(size_t)row * KNN + kk] = bj; sdw[bj] = 3.4e38f; }
    __syncthreads();
  }
}

// ---------------------------------------------------------------------------
// Fused gathered attention. One wave per query (block 256 = 8 queries).
// 4 lanes per head, 16 dims/lane. align folded to a 3-vector per (n,h):
//   align = q_h.bg_h + geo . (Wg^T q_h)
// ---------------------------------------------------------------------------
__global__ __launch_bounds__(256) void attn_kernel(
    const float* __restrict__ qf, const float* __restrict__ kf,
    const float* __restrict__ vf, const int* __restrict__ knn,
    const float* __restrict__ pos, const float* __restrict__ cptr,
    const float* __restrict__ wg, const float* __restrict__ bg,
    const float* __restrict__ ascale, const float* __restrict__ fscale,
    float* __restrict__ out) {
  __shared__ float sc[8][HEADS][KNN];
  int tid = threadIdx.x, lane = tid & 31, wid = tid >> 5;
  int n   = blockIdx.x * 8 + wid;
  int h   = lane >> 2, sub = lane & 3;
  int doff = h * HDIM + sub * 16;
  float c   = *cptr;
  float sqc = fmaxf(sqrtf(c), EPSF);

  float qv[16];
  const float4* qp = (const float4*)(qf + (size_t)n * DIMX + doff);
#pragma unroll
  for (int i = 0; i < 4; i++) {
    float4 v = qp[i];
    qv[4 * i] = v.x; qv[4 * i + 1] = v.y; qv[4 * i + 2] = v.z; qv[4 * i + 3] = v.w;
  }
  float s0 = 0.f, s1 = 0.f, s2 = 0.f, sb = 0.f;
#pragma unroll
  for (int i = 0; i < 16; i++) {
    int d = doff + i;
    s0 += qv[i] * wg[0 * DIMX + d];
    s1 += qv[i] * wg[1 * DIMX + d];
    s2 += qv[i] * wg[2 * DIMX + d];
    sb += qv[i] * bg[d];
  }
  s0 += __shfl_xor(s0, 1, 32); s0 += __shfl_xor(s0, 2, 32);
  s1 += __shfl_xor(s1, 1, 32); s1 += __shfl_xor(s1, 2, 32);
  s2 += __shfl_xor(s2, 1, 32); s2 += __shfl_xor(s2, 2, 32);
  sb += __shfl_xor(sb, 1, 32); sb += __shfl_xor(sb, 2, 32);
  float asc = ascale[h], fsc = fscale[h];

  int jk = knn[(size_t)n * KNN + lane];
  float px = pos[n * 3], py = pos[n * 3 + 1], pz = pos[n * 3 + 2];
  float p2 = px * px + py * py + pz * pz;

  for (int k = 0; k < KNN; k++) {
    int j = __shfl(jk, k, 32);
    const float4* kp = (const float4*)(kf + (size_t)j * DIMX + doff);
    float f = 0.f;
#pragma unroll
    for (int i = 0; i < 4; i++) {
      float4 v = kp[i];
      f += qv[4 * i] * v.x + qv[4 * i + 1] * v.y + qv[4 * i + 2] * v.z + qv[4 * i + 3] * v.w;
    }
    f += __shfl_xor(f, 1, 32); f += __shfl_xor(f, 2, 32);

    // geo = logmap0(mobius_add(-p_n, p_j, c), c)
    float kx = pos[j * 3], ky = pos[j * 3 + 1], kz = pos[j * 3 + 2];
    float k2 = kx * kx + ky * ky + kz * kz;
    float xy = -(px * kx + py * ky + pz * kz);
    float a  = 1.f + 2.f * c * xy + c * k2;
    float bb = 1.f - c * p2;
    float mx = -a * px + bb * kx;
    float my = -a * py + bb * ky;
    float mz = -a * pz + bb * kz;
    float invd = 1.f / fmaxf(1.f + 2.f * c * xy + c * c * p2 * k2, EPSF);
    mx *= invd; my *= invd; mz *= invd;
    float nrm   = sqrtf(mx * mx + my * my + mz * mz);
    float maxn  = (1.f - EPSF) / sqc;
    float scale = fminf(maxn / fmaxf(nrm, EPSF), 1.f);
    float yn = nrm * scale;
    float gx = 0.f, gy = 0.f, gz = 0.f;
    if (yn >= EPSF) {
      float arg = fminf(sqc * yn, 1.f - EPSF);
      float mul = (atanhf(arg) / sqc) * (scale / yn);
      gx = mx * mul; gy = my * mul; gz = mz * mul;
    }
    float align = sb + gx * s0 + gy * s1 + gz * s2;
    float score = fsc * (f * 0.125f) + asc * align;   // 1/sqrt(64)
    if (sub == 0) sc[wid][h][k] = score;
  }
  __syncthreads();

  // softmax over K (f32): 4 lanes x 8 scores per head
  float mx_ = -3.4e38f;
#pragma unroll
  for (int t = 0; t < 8; t++) mx_ = fmaxf(mx_, sc[wid][h][sub * 8 + t]);
  mx_ = fmaxf(mx_, __shfl_xor(mx_, 1, 32));
  mx_ = fmaxf(mx_, __shfl_xor(mx_, 2, 32));
  float sum = 0.f;
#pragma unroll
  for (int t = 0; t < 8; t++) {
    float e = __expf(sc[wid][h][sub * 8 + t] - mx_);
    sum += e;
    sc[wid][h][sub * 8 + t] = e;
  }
  sum += __shfl_xor(sum, 1, 32); sum += __shfl_xor(sum, 2, 32);
  float invs = 1.f / sum;
  __syncthreads();

  float accv[16];
#pragma unroll
  for (int i = 0; i < 16; i++) accv[i] = 0.f;
  for (int k = 0; k < KNN; k++) {
    int j = __shfl(jk, k, 32);
    float wgt = sc[wid][h][k] * invs;
    const float4* vp = (const float4*)(vf + (size_t)j * DIMX + doff);
#pragma unroll
    for (int i = 0; i < 4; i++) {
      float4 v = vp[i];
      accv[4 * i]     += wgt * v.x;
      accv[4 * i + 1] += wgt * v.y;
      accv[4 * i + 2] += wgt * v.z;
      accv[4 * i + 3] += wgt * v.w;
    }
  }
  float4* op = (float4*)(out + (size_t)n * DIMX + doff);
#pragma unroll
  for (int i = 0; i < 4; i++)
    op[i] = make_float4(accv[4 * i], accv[4 * i + 1], accv[4 * i + 2], accv[4 * i + 3]);
}

// ---------------------------------------------------------------------------
extern "C" void kernel_launch(void* const* d_in, const int* in_sizes, int n_in,
                              void* d_out, int out_size, void* d_ws, size_t ws_size,
                              hipStream_t stream) {
  const float* x    = (const float*)d_in[0];
  const float* pos  = (const float*)d_in[1];
  const float* c    = (const float*)d_in[2];
  const float* wq   = (const float*)d_in[3];
  const float* bq   = (const float*)d_in[4];
  const float* wk   = (const float*)d_in[5];
  const float* bk   = (const float*)d_in[6];
  const float* wv   = (const float*)d_in[7];
  const float* bv   = (const float*)d_in[8];
  const float* wg   = (const float*)d_in[9];
  const float* bg   = (const float*)d_in[10];
  const float* wo   = (const float*)d_in[11];
  const float* bo   = (const float*)d_in[12];
  const float* w1   = (const float*)d_in[13];
  const float* b1   = (const float*)d_in[14];
  const float* w2   = (const float*)d_in[15];
  const float* b2   = (const float*)d_in[16];
  const float* ln1s = (const float*)d_in[17];
  const float* ln1b = (const float*)d_in[18];
  const float* ln2s = (const float*)d_in[19];
  const float* ln2b = (const float*)d_in[20];
  const float* as_  = (const float*)d_in[21];
  const float* fs_  = (const float*)d_in[22];

  const size_t NB = (size_t)NROWS * DIMX * sizeof(float);   // 8 MB
  char* ws = (char*)d_ws;
  float* xn    = (float*)(ws + 0 * NB);
  float* qf    = (float*)(ws + 1 * NB);
  float* kf    = (float*)(ws + 2 * NB);
  float* vf    = (float*)(ws + 3 * NB);
  float* attnb = (float*)(ws + 4 * NB);
  float* x1    = (float*)(ws + 5 * NB);
  float* hbuf  = (float*)(ws + 6 * NB);
  float* tbuf  = (float*)(ws + 7 * NB);                      // 4096 x 2048
  int*   idx   = (int*)(ws + 7 * NB + (size_t)NROWS * 2048 * sizeof(float));

  // 1) LN1
  ln_rows<<<NROWS, 256, 0, stream>>>(x, ln1s, ln1b, xn);
  // 2) QKV projections (bf16 WMMA)
  dim3 g512(DIMX / 64, NROWS / 128);
  gemm_wmma_bf16<<<g512, 256, 0, stream>>>(xn, wq, bq, nullptr, qf, DIMX, DIMX, 0);
  gemm_wmma_bf16<<<g512, 256, 0, stream>>>(xn, wk, bk, nullptr, kf, DIMX, DIMX, 0);
  gemm_wmma_bf16<<<g512, 256, 0, stream>>>(xn, wv, bv, nullptr, vf, DIMX, DIMX, 0);
  // 3) hyperbolic kNN
  topk_knn<<<NROWS / 4, 128, 0, stream>>>(pos, c, idx);
  // 4) gathered attention (geo projection folded in)
  attn_kernel<<<NROWS / 8, 256, 0, stream>>>(qf, kf, vf, idx, pos, c, wg, bg,
                                             as_, fs_, attnb);
  // 5) output projection + residual
  gemm_wmma_bf16<<<g512, 256, 0, stream>>>(attnb, wo, bo, x, x1, DIMX, DIMX, 1);
  // 6) LN2
  ln_rows<<<NROWS, 256, 0, stream>>>(x1, ln2s, ln2b, hbuf);
  // 7) MLP
  dim3 g2048(2048 / 64, NROWS / 128);
  gemm_wmma_bf16<<<g2048, 256, 0, stream>>>(hbuf, w1, b1, nullptr, tbuf, DIMX, 2048, 2);
  gemm_wmma_bf16<<<g512, 256, 0, stream>>>(tbuf, w2, b2, x1, (float*)d_out, 2048, DIMX, 1);
}
